// AGTBlock_29059748725684
// MI455X (gfx1250) — compile-verified, matrix operand order checked by reading, not run on hardware
//
#include <hip/hip_runtime.h>
#include <hip/hip_bf16.h>
#include <math.h>

// ---------------------------------------------------------------------------
// AGT block for MI455X (gfx1250, wave32, WMMA 16x16x32 f16, async LDS copies,
// double-buffered GEMM, persistent LDS-resident edge-MLP weight)
// ---------------------------------------------------------------------------

typedef __attribute__((ext_vector_type(16))) _Float16 v16h;
typedef __attribute__((ext_vector_type(8)))  _Float16 v8h;
typedef __attribute__((ext_vector_type(8)))  float    v8f;

#define DD 256

// ------------------------- ordered-uint float max ---------------------------
__device__ __forceinline__ unsigned f2ord(float f) {
    unsigned u = __float_as_uint(f);
    return (u & 0x80000000u) ? ~u : (u | 0x80000000u);
}
__device__ __forceinline__ float ord2f(unsigned u) {
    return (u & 0x80000000u) ? __uint_as_float(u & 0x7fffffffu)
                             : __uint_as_float(~u);
}

// -------------------- CDNA5 async global->LDS copy (b128) -------------------
// GVS mode: mem = SADDR(64) + VADDR(32) + IOFFSET ; LDS addr from VGPR.
__device__ __forceinline__ void async_b128(unsigned lds_off, unsigned gvoff,
                                           unsigned long long base) {
    asm volatile("global_load_async_to_lds_b128 %0, %1, %2"
                 :: "v"(lds_off), "v"(gvoff), "s"(base) : "memory");
}
__device__ __forceinline__ void wait_async0() {
    asm volatile("s_wait_asynccnt 0x0" ::: "memory");
}
// previous stage (3 in-flight copies belong to the *next* stage)
__device__ __forceinline__ void wait_async3() {
    asm volatile("s_wait_asynccnt 0x3" ::: "memory");
}

// fragment: two contiguous 8xf16 runs (k..k+7, k+16..k+23) -> one v16h
__device__ __forceinline__ v16h frag_load(const char* p /*16B aligned*/) {
    v8h lo = *(const v8h*)(p);
    v8h hi = *(const v8h*)(p + 32);
    return __builtin_shufflevector(lo, hi, 0, 1, 2, 3, 4, 5, 6, 7,
                                           8, 9, 10, 11, 12, 13, 14, 15);
}

// ---------------------------------------------------------------------------
// GEMM: Out[rows,256] = act(Ah[rows,256] @ W + bias), W given TRANSPOSED f16
// block: 256 threads = 8 waves; block tile 128 rows x 64 cols
// double-buffered dynamic LDS:
//   buf = As 128x40 f16 (10240 B) | Wt 64x40 f16 (5120 B) = 15360 B; 2 bufs
// ---------------------------------------------------------------------------
#define GEMM_BUF 15360
#define GEMM_LDS (2 * GEMM_BUF)
#define AS_STRIDE 80   // bytes per LDS row (40 f16)
#define WT_OFF 10240

__device__ __forceinline__ void gemm_stage(
    int t, int rowBase, int col0, int rows, int k0, unsigned bufBase,
    unsigned long long aBase, unsigned long long wBase)
{
    // A tile: 128 rows x 64 B  (2 x b128 per thread)
#pragma unroll
    for (int i = 0; i < 2; ++i) {
        int idx = t + i * 256;
        int r = idx >> 2, seg = idx & 3;
        int grow = rowBase + r;
        grow = (grow < rows) ? grow : (rows - 1);
        async_b128(bufBase + (unsigned)(r * AS_STRIDE + seg * 16),
                   (unsigned)grow * 512u + (unsigned)k0 * 2u +
                       (unsigned)seg * 16u,
                   aBase);
    }
    // W^T tile: 64 rows x 64 B (1 x b128 per thread)
    {
        int r = t >> 2, seg = t & 3;
        async_b128(bufBase + (unsigned)(WT_OFF + r * AS_STRIDE + seg * 16),
                   (unsigned)(col0 + r) * 512u + (unsigned)k0 * 2u +
                       (unsigned)seg * 16u,
                   wBase);
    }
}

template <bool RELU, bool F32OUT, bool F16OUT>
__global__ __launch_bounds__(256) void gemm_bias_act(
    const _Float16* __restrict__ Ah, const _Float16* __restrict__ WT,
    const float* __restrict__ bias, float* __restrict__ Out,
    _Float16* __restrict__ OutH, int rows)
{
    extern __shared__ char smem[];

    const int t    = threadIdx.x;
    const int lane = t & 31;
    const int wave = t >> 5;
    const int rowBase = blockIdx.x * 128;
    const int col0    = blockIdx.y * 64;

    const int lrow = lane & 15;
    const int kb   = (lane >> 4) << 3;   // 0 or 8

    const unsigned long long aBase = (unsigned long long)Ah;
    const unsigned long long wBase = (unsigned long long)WT;

    v8f acc[4] = {};

    // prestage k-step 0 into buffer 0
    gemm_stage(t, rowBase, col0, rows, 0, 0u, aBase, wBase);

    for (int k0 = 0; k0 < 256; k0 += 32) {
        const unsigned cur = ((k0 >> 5) & 1) ? (unsigned)GEMM_BUF : 0u;
        const unsigned nxt = (unsigned)GEMM_BUF - cur;
        const bool hasNext = (k0 + 32) < 256;

        if (hasNext) {
            // overlap: issue next stage's copies before waiting on current
            gemm_stage(t, rowBase, col0, rows, k0 + 32, nxt, aBase, wBase);
            wait_async3();
        } else {
            wait_async0();
        }
        __syncthreads();

        v16h a = frag_load(smem + cur + (wave * 16 + lrow) * AS_STRIDE + kb * 2);
#pragma unroll
        for (int nt = 0; nt < 4; ++nt) {
            v16h b = frag_load(smem + cur + WT_OFF +
                               (nt * 16 + lrow) * AS_STRIDE + kb * 2);
            acc[nt] = __builtin_amdgcn_wmma_f32_16x16x32_f16(
                false, a, false, b, (short)0, acc[nt], false, false);
        }
        __syncthreads();
    }

    // ---- epilogue ----
    const int hi = (lane >= 16) ? 8 : 0;
#pragma unroll
    for (int nt = 0; nt < 4; ++nt) {
        int col = col0 + nt * 16 + lrow;
        float bv = bias[col];
#pragma unroll
        for (int r = 0; r < 8; ++r) {
            int row = rowBase + wave * 16 + r + hi;
            if (row < rows) {
                float v = acc[nt][r] + bv;
                if (RELU) v = fmaxf(v, 0.0f);
                if (F32OUT) Out[(size_t)row * DD + col] = v;
                if (F16OUT) OutH[(size_t)row * DD + col] = (_Float16)v;
            }
        }
    }
}

// ---------------------------------------------------------------------------
// conversion helpers
// ---------------------------------------------------------------------------
__global__ __launch_bounds__(256) void cvt_f16(
    const float* __restrict__ in, _Float16* __restrict__ out, int n)
{
    int i = blockIdx.x * 256 + threadIdx.x;
    if (i < n) out[i] = (_Float16)in[i];
}

// WT[n][k] = (f16) W[k][n]   (256x256)
__global__ __launch_bounds__(256) void cvt_transpose_w(
    const float* __restrict__ W, _Float16* __restrict__ WT)
{
    int n = blockIdx.x, k = threadIdx.x;
    WT[n * DD + k] = (_Float16)W[k * DD + n];
}

// ---------------------------------------------------------------------------
// init: zero softmax state and output accumulator
// ---------------------------------------------------------------------------
__global__ __launch_bounds__(256) void init_state(
    unsigned* __restrict__ amax, float* __restrict__ denom,
    float* __restrict__ accum, int Nn, int ND)
{
    int i = blockIdx.x * 256 + threadIdx.x;
    if (i < ND) accum[i] = 0.0f;
    if (i < Nn) { amax[i] = 0u; denom[i] = 0.0f; }
}

// ---------------------------------------------------------------------------
// alpha[e] = dot(Q[dst],K[src]) / 16 ; amax[dst] = max   (wave per edge)
// ---------------------------------------------------------------------------
__global__ __launch_bounds__(256) void edge_alpha(
    const float* __restrict__ Q, const float* __restrict__ Kp,
    const int* __restrict__ ei, float* __restrict__ alpha,
    unsigned* __restrict__ amax, int Ee)
{
    int e    = blockIdx.x * 8 + (threadIdx.x >> 5);
    int lane = threadIdx.x & 31;
    if (e >= Ee) return;
    int s = ei[e];
    int d = ei[Ee + e];
    const float4* q4 = (const float4*)(Q + (size_t)d * DD) + lane * 2;
    const float4* k4 = (const float4*)(Kp + (size_t)s * DD) + lane * 2;
    float4 a0 = q4[0], a1 = q4[1], b0 = k4[0], b1 = k4[1];
    float acc = a0.x * b0.x + a0.y * b0.y + a0.z * b0.z + a0.w * b0.w +
                a1.x * b1.x + a1.y * b1.y + a1.z * b1.z + a1.w * b1.w;
#pragma unroll
    for (int off = 16; off > 0; off >>= 1)
        acc += __shfl_xor(acc, off, 32);
    if (lane == 0) {
        float a = acc * (1.0f / 16.0f);   // 1/sqrt(256)
        alpha[e] = a;
        atomicMax(&amax[d], f2ord(a));
    }
}

// ---------------------------------------------------------------------------
// ex[e] = exp(alpha[e]-amax[dst]); denom[dst] += ex
// ---------------------------------------------------------------------------
__global__ __launch_bounds__(256) void edge_exp(
    const float* __restrict__ alpha, const unsigned* __restrict__ amax,
    const int* __restrict__ ei, float* __restrict__ exv,
    float* __restrict__ denom, int Ee)
{
    int e = blockIdx.x * 256 + threadIdx.x;
    if (e >= Ee) return;
    int d = ei[Ee + e];
    float ex = __expf(alpha[e] - ord2f(amax[d]));
    exv[e] = ex;
    atomicAdd(&denom[d], ex);
}

// ---------------------------------------------------------------------------
// Persistent fused position-MLP + message + scatter.
// The full Wp2^T (256x256 f16) lives in LDS for the whole kernel; blocks
// grid-stride over 16-edge tiles.  MI455X: 320 KB LDS/WGP -> 2 blocks/WGP.
// dynamic LDS layout (bytes):
//   Wt2  @ 0      : 256 rows x 528 B stride (512 B data + 16 pad) = 135168
//   Hs   @ 135168 : 16 x 264 f16 (row stride 528)                 = 8448
//   srcS @ 143616 : 16 int ; dstS @ 143680 ; attn @ 143744 ; rel @ 143808
//   total 144000
// ---------------------------------------------------------------------------
#define PM_LDS 144000
#define PM_WT_STRIDE 528
#define PM_HS_OFF 135168
#define PM_HS_STRIDE 528
#define PM_GRID 1024

__global__ __launch_bounds__(256) void pos_msg(
    const float* __restrict__ pos, const int* __restrict__ ei,
    const float* __restrict__ Wp1, const float* __restrict__ bp1,
    const _Float16* __restrict__ Wp2T, const float* __restrict__ bp2,
    const float* __restrict__ V,  const float* __restrict__ exv,
    const float* __restrict__ denom, float* __restrict__ accum, int Ee)
{
    extern __shared__ char smem[];
    _Float16* Hs    = (_Float16*)(smem + PM_HS_OFF);
    int*      srcS  = (int*)(smem + 143616);
    int*      dstS  = (int*)(smem + 143680);
    float*    attnS = (float*)(smem + 143744);
    float*    relS  = (float*)(smem + 143808);   // [16][3]

    const int t    = threadIdx.x;
    const int lane = t & 31;
    const int wave = t >> 5;
    const int lrow = lane & 15;
    const int kb   = (lane >> 4) << 3;

    // ---- one-time: async-load full Wp2^T into LDS (32 x b128 per thread) ---
    {
        const unsigned long long wBase = (unsigned long long)Wp2T;
#pragma unroll
        for (int i = 0; i < 32; ++i) {
            int idx = t + i * 256;
            int r = idx >> 5, seg = idx & 31;          // row 0..255, 16B seg
            async_b128((unsigned)(r * PM_WT_STRIDE + seg * 16),
                       (unsigned)(r * 512 + seg * 16), wBase);
        }
        wait_async0();
        __syncthreads();
    }

    const int nTiles = (Ee + 15) / 16;
    for (int tile = blockIdx.x; tile < nTiles; tile += gridDim.x) {
        const int e0 = tile * 16;
        __syncthreads();   // previous tile's readers done before Hs overwrite

        if (t < 16) {
            int e = e0 + t;
            int s_ = 0, d_ = 0;
            float at = 0.0f, r0 = 0.0f, r1 = 0.0f, r2 = 0.0f;
            if (e < Ee) {
                s_ = ei[e];
                d_ = ei[Ee + e];
                r0 = pos[s_ * 3 + 0] - pos[d_ * 3 + 0];
                r1 = pos[s_ * 3 + 1] - pos[d_ * 3 + 1];
                r2 = pos[s_ * 3 + 2] - pos[d_ * 3 + 2];
                at = exv[e] / (denom[d_] + 1e-16f);
            }
            srcS[t] = s_; dstS[t] = d_; attnS[t] = at;
            relS[t * 3 + 0] = r0; relS[t * 3 + 1] = r1; relS[t * 3 + 2] = r2;
        }
        __syncthreads();

        // layer 1 (K=3): thread t computes column t for all 16 edges
        {
            float w0 = Wp1[t], w1 = Wp1[DD + t], w2 = Wp1[2 * DD + t];
            float bb = bp1[t];
#pragma unroll
            for (int i = 0; i < 16; ++i) {
                float hv = relS[i * 3 + 0] * w0 + relS[i * 3 + 1] * w1 +
                           relS[i * 3 + 2] * w2 + bb;
                Hs[i * 264 + t] = (_Float16)fmaxf(hv, 0.0f);
            }
        }
        __syncthreads();

        // barrier-free k-loop: Hs + LDS-resident Wt2 -> 16 WMMAs
        v8f acc[2] = {};
#pragma unroll
        for (int k0 = 0; k0 < 256; k0 += 32) {
            v16h a = frag_load(smem + PM_HS_OFF + lrow * PM_HS_STRIDE +
                               (k0 + kb) * 2);
#pragma unroll
            for (int nt = 0; nt < 2; ++nt) {
                v16h b = frag_load(smem +
                                   (wave * 32 + nt * 16 + lrow) * PM_WT_STRIDE +
                                   (k0 + kb) * 2);
                acc[nt] = __builtin_amdgcn_wmma_f32_16x16x32_f16(
                    false, a, false, b, (short)0, acc[nt], false, false);
            }
        }

        // epilogue: scatter-add attn*(V[src]+pe) into accum[dst]
        const int hi = (lane >= 16) ? 8 : 0;
#pragma unroll
        for (int nt = 0; nt < 2; ++nt) {
            int col = wave * 32 + nt * 16 + lrow;
            float bv = bp2[col];
#pragma unroll
            for (int r = 0; r < 8; ++r) {
                int eloc = r + hi;
                if (e0 + eloc < Ee) {
                    float pe  = acc[nt][r] + bv;
                    float msg = attnS[eloc] *
                                (V[(size_t)srcS[eloc] * DD + col] + pe);
                    atomicAdd(&accum[(size_t)dstS[eloc] * DD + col], msg);
                }
            }
        }
    }
}

// ---------------------------------------------------------------------------
// out = LN( LN(accum+h)*g1+b1 + x )*g2+b2   (wave per row, 8 rows/block)
// ---------------------------------------------------------------------------
__global__ __launch_bounds__(256) void fused_double_ln(
    const float* __restrict__ accum, const float* __restrict__ h,
    const float* __restrict__ x,
    const float* __restrict__ g1, const float* __restrict__ b1,
    const float* __restrict__ g2, const float* __restrict__ b2,
    float* __restrict__ out, int Nn)
{
    int row  = blockIdx.x * 8 + (threadIdx.x >> 5);
    int lane = threadIdx.x & 31;
    if (row >= Nn) return;
    size_t base = (size_t)row * DD;

    float v[8];
    float s = 0.0f;
#pragma unroll
    for (int i = 0; i < 8; ++i) {
        int c = lane + i * 32;
        v[i] = accum[base + c] + h[base + c];
        s += v[i];
    }
#pragma unroll
    for (int off = 16; off > 0; off >>= 1) s += __shfl_xor(s, off, 32);
    float mu = s * (1.0f / 256.0f);
    float vs = 0.0f;
#pragma unroll
    for (int i = 0; i < 8; ++i) { float d = v[i] - mu; vs += d * d; }
#pragma unroll
    for (int off = 16; off > 0; off >>= 1) vs += __shfl_xor(vs, off, 32);
    float rs = rsqrtf(vs * (1.0f / 256.0f) + 1e-5f);

    float s2 = 0.0f;
#pragma unroll
    for (int i = 0; i < 8; ++i) {
        int c = lane + i * 32;
        float hn = (v[i] - mu) * rs * g1[c] + b1[c];
        v[i] = hn + x[base + c];
        s2 += v[i];
    }
#pragma unroll
    for (int off = 16; off > 0; off >>= 1) s2 += __shfl_xor(s2, off, 32);
    float mu2 = s2 * (1.0f / 256.0f);
    float vs2 = 0.0f;
#pragma unroll
    for (int i = 0; i < 8; ++i) { float d = v[i] - mu2; vs2 += d * d; }
#pragma unroll
    for (int off = 16; off > 0; off >>= 1) vs2 += __shfl_xor(vs2, off, 32);
    float rs2 = rsqrtf(vs2 * (1.0f / 256.0f) + 1e-5f);

#pragma unroll
    for (int i = 0; i < 8; ++i) {
        int c = lane + i * 32;
        out[base + c] = (v[i] - mu2) * rs2 * g2[c] + b2[c];
    }
}

// ---------------------------------------------------------------------------
// launcher
// ---------------------------------------------------------------------------
extern "C" void kernel_launch(void* const* d_in, const int* in_sizes, int n_in,
                              void* d_out, int out_size, void* d_ws, size_t ws_size,
                              hipStream_t stream)
{
    const float* x    = (const float*)d_in[0];
    const int*   ei   = (const int*)d_in[1];
    const float* pos  = (const float*)d_in[2];
    const float* Wm1  = (const float*)d_in[3];
    const float* bm1  = (const float*)d_in[4];
    const float* Wm2  = (const float*)d_in[5];
    const float* bm2  = (const float*)d_in[6];
    const float* Wq   = (const float*)d_in[7];
    const float* bq   = (const float*)d_in[8];
    const float* Wk   = (const float*)d_in[9];
    const float* bk   = (const float*)d_in[10];
    const float* Wv   = (const float*)d_in[11];
    const float* bv   = (const float*)d_in[12];
    const float* Wp1  = (const float*)d_in[13];
    const float* bp1  = (const float*)d_in[14];
    const float* Wp2  = (const float*)d_in[15];
    const float* bp2  = (const float*)d_in[16];
    const float* g1   = (const float*)d_in[17];
    const float* b1n  = (const float*)d_in[18];
    const float* g2   = (const float*)d_in[19];
    const float* b2n  = (const float*)d_in[20];

    const int Nn = in_sizes[0] / DD;       // 10000
    const int Ee = in_sizes[1] / 2;        // 320000
    const size_t nd = (size_t)Nn * DD;
    const int ND = (int)nd;

    // ---- workspace carve (all 16B aligned) ----
    char* w = (char*)d_ws;
    float* accum_ = (float*)w;           w += nd * 4;
    float* h_     = (float*)w;           w += nd * 4;
    float* Q_     = (float*)w;           w += nd * 4;
    float* K_     = (float*)w;           w += nd * 4;
    float* V_     = (float*)w;           w += nd * 4;
    float* alpha_ = (float*)w;           w += (size_t)Ee * 4;
    float* ex_    = (float*)w;           w += (size_t)Ee * 4;
    float* denom_ = (float*)w;           w += (size_t)Nn * 4;
    unsigned* amax_ = (unsigned*)w;      w += (size_t)Nn * 4;
    _Float16* xh  = (_Float16*)w;        w += nd * 2;
    _Float16* th  = (_Float16*)w;        w += nd * 2;
    _Float16* hh  = (_Float16*)w;        w += nd * 2;
    _Float16* WTm1 = (_Float16*)w;       w += (size_t)DD * DD * 2;
    _Float16* WTm2 = (_Float16*)w;       w += (size_t)DD * DD * 2;
    _Float16* WTq  = (_Float16*)w;       w += (size_t)DD * DD * 2;
    _Float16* WTk  = (_Float16*)w;       w += (size_t)DD * DD * 2;
    _Float16* WTv  = (_Float16*)w;       w += (size_t)DD * DD * 2;
    _Float16* WTp2 = (_Float16*)w;       w += (size_t)DD * DD * 2;
    float* out    = (float*)d_out;

    dim3 blk(256);
    dim3 gemmGrid((Nn + 127) / 128, 4);

    // ---- conversions (f16 activations, transposed f16 weights) ----
    cvt_f16<<<(ND + 255) / 256, blk, 0, stream>>>(x, xh, ND);
    cvt_transpose_w<<<DD, blk, 0, stream>>>(Wm1, WTm1);
    cvt_transpose_w<<<DD, blk, 0, stream>>>(Wm2, WTm2);
    cvt_transpose_w<<<DD, blk, 0, stream>>>(Wq,  WTq);
    cvt_transpose_w<<<DD, blk, 0, stream>>>(Wk,  WTk);
    cvt_transpose_w<<<DD, blk, 0, stream>>>(Wv,  WTv);
    cvt_transpose_w<<<DD, blk, 0, stream>>>(Wp2, WTp2);

    // ---- node MLP + projections (WMMA, double-buffered async staging) ----
    gemm_bias_act<true,  false, true ><<<gemmGrid, blk, GEMM_LDS, stream>>>(
        xh, WTm1, bm1, nullptr, th, Nn);
    gemm_bias_act<false, true,  true ><<<gemmGrid, blk, GEMM_LDS, stream>>>(
        th, WTm2, bm2, h_, hh, Nn);
    gemm_bias_act<false, true,  false><<<gemmGrid, blk, GEMM_LDS, stream>>>(
        hh, WTq, bq, Q_, nullptr, Nn);
    gemm_bias_act<false, true,  false><<<gemmGrid, blk, GEMM_LDS, stream>>>(
        hh, WTk, bk, K_, nullptr, Nn);
    gemm_bias_act<false, true,  false><<<gemmGrid, blk, GEMM_LDS, stream>>>(
        hh, WTv, bv, V_, nullptr, Nn);

    // ---- softmax state + accumulator ----
    init_state<<<(ND + 255) / 256, blk, 0, stream>>>(amax_, denom_, accum_, Nn, ND);

    // ---- attention scores + segment softmax ----
    edge_alpha<<<(Ee + 7) / 8, blk, 0, stream>>>(Q_, K_, ei, alpha_, amax_, Ee);
    edge_exp<<<(Ee + 255) / 256, blk, 0, stream>>>(alpha_, amax_, ei, ex_, denom_, Ee);

    // ---- persistent fused position MLP + weighted message scatter (WMMA) ---
    int nTiles = (Ee + 15) / 16;
    int pmGrid = (nTiles < PM_GRID) ? nTiles : PM_GRID;
    pos_msg<<<pmGrid, blk, PM_LDS, stream>>>(
        pos, ei, Wp1, bp1, WTp2, bp2, V_, ex_, denom_, accum_, Ee);

    // ---- residual + two layer norms ----
    fused_double_ln<<<(Nn + 7) / 8, blk, 0, stream>>>(accum_, h_, x,
                                                      g1, b1n, g2, b2n, out, Nn);
}
